// Pointnet2Backbone_54056458388089
// MI455X (gfx1250) — matrix operand order repaired and checked by reading, not compiled, not run on hardware
//
#include <hip/hip_runtime.h>
#include <hip/hip_fp16.h>

typedef __attribute__((ext_vector_type(16))) _Float16 v16h;
typedef __attribute__((ext_vector_type(8)))  float    v8f;

#define WGS 256
#define LDSA_STRIDE 288   // max padded Cin for SA stages: 259 -> 288; also covers Cout<=256
#define FP_STR0 512
#define FP_STR1 272

// ---------------------------------------------------------------------------
// Pack an f32 weight matrix (Cin x Cout, row-major) into f16 WMMA B-fragment
// order, zero-padded so K covers ceil(Cin/32)*32:
//   Wp[((kt*nCt + ct)*32 + lane)*16 + h] = W[(kt*32 + (lane>>4)*16 + h)*Cout
//                                            + ct*16 + (lane&15)]
// Each lane's fragment becomes 32 contiguous bytes -> one v16h load in GEMM.
// ---------------------------------------------------------------------------
__global__ void k_pack_w(const float* __restrict__ src, _Float16* __restrict__ dst,
                         int Cin, int Cout, int total) {
  int t = blockIdx.x * blockDim.x + threadIdx.x;
  if (t >= total) return;
  const int nCt = Cout >> 4;
  int h   = t & 15;
  int l   = (t >> 4) & 31;
  int rem = t >> 9;
  int ct  = rem % nCt;
  int kt  = rem / nCt;
  int k   = kt * 32 + (l >> 4) * 16 + h;
  int n   = ct * 16 + (l & 15);
  dst[t] = (k < Cin) ? (_Float16)src[(size_t)k * Cout + n] : (_Float16)0.0f;
}

// ---------------------------------------------------------------------------
// Farthest-point sampling: one workgroup (1024 thr = 32 waves) per batch.
// ---------------------------------------------------------------------------
__global__ __launch_bounds__(1024) void k_fps(const float* __restrict__ xyz, int N, int npoint,
                                              int* __restrict__ out, float* __restrict__ dist) {
  const int b = blockIdx.x;
  const float* X = xyz + (size_t)b * N * 3;
  float* D = dist + (size_t)b * N;
  __shared__ float s_val[32];
  __shared__ int   s_idx[32];
  __shared__ int   s_last;
  const int tid = threadIdx.x;
  for (int i = tid; i < N; i += blockDim.x) D[i] = 1e10f;
  if (tid == 0) { s_last = 0; out[(size_t)b * npoint] = 0; }
  __syncthreads();
  for (int it = 1; it < npoint; ++it) {
    const int last = s_last;
    const float px = X[last*3+0], py = X[last*3+1], pz = X[last*3+2];
    float best = -1.0f; int bi = 0;
    for (int i = tid; i < N; i += blockDim.x) {
      float dx = X[i*3+0]-px, dy = X[i*3+1]-py, dz = X[i*3+2]-pz;
      float d  = dx*dx + dy*dy + dz*dz;
      float dm = fminf(D[i], d);
      D[i] = dm;
      if (dm > best) { best = dm; bi = i; }
    }
    #pragma unroll
    for (int off = 16; off > 0; off >>= 1) {   // wave32 argmax
      float ov = __shfl_down(best, off);
      int   oi = __shfl_down(bi,   off);
      if (ov > best) { best = ov; bi = oi; }
    }
    const int wave = tid >> 5, lane = tid & 31;
    if (lane == 0) { s_val[wave] = best; s_idx[wave] = bi; }
    __syncthreads();
    if (tid < 32) {
      best = s_val[tid]; bi = s_idx[tid];
      #pragma unroll
      for (int off = 16; off > 0; off >>= 1) {
        float ov = __shfl_down(best, off);
        int   oi = __shfl_down(bi,   off);
        if (ov > best) { best = ov; bi = oi; }
      }
      if (tid == 0) { out[(size_t)b * npoint + it] = bi; s_last = bi; }
    }
    __syncthreads();
  }
}

// new_xyz[b,s] = xyz[b, fps_idx[b,s]]
__global__ void k_gather_xyz(const float* __restrict__ xyz, int N,
                             const int* __restrict__ fidx, int S,
                             float* __restrict__ nxyz, int BS) {
  int t = blockIdx.x * blockDim.x + threadIdx.x;
  if (t >= BS) return;
  int b = t / S;
  int pi = fidx[t];
  const float* p = xyz + ((size_t)b * N + pi) * 3;
  float* o = nxyz + (size_t)t * 3;
  o[0] = p[0]; o[1] = p[1]; o[2] = p[2];
}

// Ball query: first K in-radius points in ascending index order, pad with first.
__global__ void k_ball_query(const float* __restrict__ xyz, int N,
                             const float* __restrict__ nxyz, int S, int K,
                             float r2, int* __restrict__ idx, int BS) {
  int t = blockIdx.x * blockDim.x + threadIdx.x;
  if (t >= BS) return;
  int b = t / S;
  const float* c = nxyz + (size_t)t * 3;
  const float* X = xyz + (size_t)b * N * 3;
  const float px = c[0], py = c[1], pz = c[2];
  int* o = idx + (size_t)t * K;
  int cnt = 0, first = -1;
  for (int i = 0; i < N && cnt < K; ++i) {
    float dx = X[i*3+0]-px, dy = X[i*3+1]-py, dz = X[i*3+2]-pz;
    if (dx*dx + dy*dy + dz*dz < r2) { if (first < 0) first = i; o[cnt++] = i; }
  }
  if (first < 0) first = N - 1;
  for (; cnt < K; ++cnt) o[cnt] = first;
}

// 3-NN with inverse-distance weights (matches top_k(-d2,3) + 1/(d2+1e-8)).
__global__ void k_knn3(const float* __restrict__ unk, int Nn,
                       const float* __restrict__ kn, int M,
                       int* __restrict__ oi, float* __restrict__ ow, int BN) {
  int t = blockIdx.x * blockDim.x + threadIdx.x;
  if (t >= BN) return;
  int b = t / Nn;
  const float* u  = unk + (size_t)t * 3;
  const float* Kx = kn  + (size_t)b * M * 3;
  float d0 = 1e30f, d1 = 1e30f, d2 = 1e30f; int i0 = 0, i1 = 0, i2 = 0;
  const float ux = u[0], uy = u[1], uz = u[2];
  for (int i = 0; i < M; ++i) {
    float dx = Kx[i*3+0]-ux, dy = Kx[i*3+1]-uy, dz = Kx[i*3+2]-uz;
    float d = dx*dx + dy*dy + dz*dz;
    if (d < d0)      { d2=d1; i2=i1; d1=d0; i1=i0; d0=d; i0=i; }
    else if (d < d1) { d2=d1; i2=i1; d1=d; i1=i; }
    else if (d < d2) { d2=d; i2=i; }
  }
  float w0 = 1.0f/(d0+1e-8f), w1 = 1.0f/(d1+1e-8f), w2 = 1.0f/(d2+1e-8f);
  float s = w0 + w1 + w2;
  oi[(size_t)t*3+0] = i0; oi[(size_t)t*3+1] = i1; oi[(size_t)t*3+2] = i2;
  ow[(size_t)t*3+0] = w0/s; ow[(size_t)t*3+1] = w1/s; ow[(size_t)t*3+2] = w2/s;
}

// ---------------------------------------------------------------------------
// Fragment loaders.
// ---------------------------------------------------------------------------
__device__ __forceinline__ v16h load_a_frag(const _Float16* sA, int strA,
                                            int row, int kk, int half16) {
  // halves 0..15 -> K=kk+half16*16+h (contiguous, matches k_pack_w)
  const _Float16* ap = sA + (size_t)row * strA + kk + half16 * 16;
  v16h a;
  #pragma unroll
  for (int h = 0; h < 8; ++h) { a[h] = ap[h]; a[h + 8] = ap[8 + h]; }
  return a;
}
__device__ __forceinline__ v16h load_b_frag(const _Float16* __restrict__ Wp,
                                            int nCt, int kt, int ct, int lane) {
  return *(const v16h*)(Wp + (((size_t)kt * nCt + ct) * 32 + lane) * 16);
}

// ---------------------------------------------------------------------------
// WMMA GEMM layer over a 64-row LDS-resident tile.
// 8 waves = 4 row-tiles x 2 col phases; each pass computes TWO 16-col tiles
// with independent accumulators (Cout is a multiple of 64 -> no guards), so A
// is loaded once per K-step for 2 WMMAs. The K loop is software-pipelined:
// next A/B fragments are issued before the current WMMAs so waits are partial.
// ---------------------------------------------------------------------------
__device__ __forceinline__ void wmma_mlp_layer(
    const _Float16* sA, int strA,
    _Float16* sO, int strO,
    const _Float16* __restrict__ Wp,  // packed fragments (see k_pack_w)
    const float* __restrict__ bias,   // (Cout) f32
    int Cin, int Cout, int lane, int wave)
{
  const int m0     = (wave >> 1) * 16;
  const int half16 = lane >> 4;
  const int l15    = lane & 15;
  const int nCt    = Cout >> 4;
  const int row    = m0 + l15;
  // col tiles owned by this wave: ct = (wave&1) + 2*j, j in [0, nCt/2);
  // processed in pairs (j, j+1); nCt/2 is even for Cout in {64,128,256}.
  for (int j = 0; j < (nCt >> 1); j += 2) {
    const int ct0 = (wave & 1) + 2 * j;
    const int ct1 = ct0 + 2;
    v8f acc0 = {}, acc1 = {};
    v16h a  = load_a_frag(sA, strA, row, 0, half16);
    v16h b0 = load_b_frag(Wp, nCt, 0, ct0, lane);
    v16h b1 = load_b_frag(Wp, nCt, 0, ct1, lane);
    for (int kk = 32; kk < Cin; kk += 32) {
      v16h an  = load_a_frag(sA, strA, row, kk, half16);
      v16h b0n = load_b_frag(Wp, nCt, kk >> 5, ct0, lane);
      v16h b1n = load_b_frag(Wp, nCt, kk >> 5, ct1, lane);
      acc0 = __builtin_amdgcn_wmma_f32_16x16x32_f16(false, a, false, b0,
                                                    (short)0, acc0, false, false);
      acc1 = __builtin_amdgcn_wmma_f32_16x16x32_f16(false, a, false, b1,
                                                    (short)0, acc1, false, false);
      a = an; b0 = b0n; b1 = b1n;
    }
    acc0 = __builtin_amdgcn_wmma_f32_16x16x32_f16(false, a, false, b0,
                                                  (short)0, acc0, false, false);
    acc1 = __builtin_amdgcn_wmma_f32_16x16x32_f16(false, a, false, b1,
                                                  (short)0, acc1, false, false);
    const int n0 = ct0 * 16 + l15;
    const int n1 = ct1 * 16 + l15;
    const float bv0 = bias[n0];
    const float bv1 = bias[n1];
    #pragma unroll
    for (int v = 0; v < 8; ++v) {   // D row = m0 + half16*8 + v
      const int r = m0 + half16 * 8 + v;
      float x0 = fmaxf(acc0[v] + bv0, 0.0f);   // ReLU (all layers)
      float x1 = fmaxf(acc1[v] + bv1, 0.0f);
      sO[(size_t)r * strO + n0] = (_Float16)x0;
      sO[(size_t)r * strO + n1] = (_Float16)x1;
    }
  }
}

// ---------------------------------------------------------------------------
// Fused SA stage: gather(+normalize, +concat feats) -> MLP x3 -> max over K.
// 64 rows per workgroup = G groups of K neighbors (G = 64/K); 256 threads.
// Activations never leave LDS; only (G x C2) reduced features hit HBM.
// ---------------------------------------------------------------------------
__global__ __launch_bounds__(256) void k_sa_mlp(
    const float* __restrict__ xyz, int N,
    const float* __restrict__ nxyz, int S,
    const int* __restrict__ nbr, int K,
    const float* __restrict__ fin, int Cf,
    float inv_r,
    const _Float16* __restrict__ w0, const float* __restrict__ b0, int C0,
    const _Float16* __restrict__ w1, const float* __restrict__ b1, int C1,
    const _Float16* __restrict__ w2, const float* __restrict__ b2, int C2,
    float* __restrict__ fout)
{
  __shared__ _Float16 sA[64 * LDSA_STRIDE];
  __shared__ _Float16 sB[64 * LDSA_STRIDE];
  __shared__ int   s_pi[64];
  __shared__ float s_ctr[4][3];
  const int b   = blockIdx.y;
  const int G   = 64 / K;
  const int s0  = blockIdx.x * G;
  const int tid = threadIdx.x;
  const int lane = tid & 31, wave = tid >> 5;
  const int Cin  = 3 + Cf;
  const int Cinp = (Cin + 31) & ~31;

  if (tid < 64) {
    int g = tid / K, k = tid - g * K;
    s_pi[tid] = nbr[((size_t)b * S + (s0 + g)) * K + k];
  }
  if (tid < G * 3) {
    int g = tid / 3, c = tid - g * 3;
    s_ctr[g][c] = nxyz[((size_t)b * S + (s0 + g)) * 3 + c];
  }
  __syncthreads();
  // Stage gathered rows (normalized xyz ++ features), zero-padded to Cinp.
  const float* Xb = xyz + (size_t)b * N * 3;
  for (int u = tid; u < 64 * Cinp; u += WGS) {
    int r = u / Cinp, c = u - r * Cinp;
    int g = r / K;
    int pi = s_pi[r];
    float v = 0.0f;
    if (c < 3)        v = (Xb[pi * 3 + c] - s_ctr[g][c]) * inv_r;
    else if (c < Cin) v = fin[((size_t)b * N + pi) * Cf + (c - 3)];
    sA[(size_t)r * LDSA_STRIDE + c] = (_Float16)v;
  }
  __syncthreads();
  wmma_mlp_layer(sA, LDSA_STRIDE, sB, LDSA_STRIDE, w0, b0, Cin, C0, lane, wave);
  __syncthreads();
  wmma_mlp_layer(sB, LDSA_STRIDE, sA, LDSA_STRIDE, w1, b1, C0, C1, lane, wave);
  __syncthreads();
  wmma_mlp_layer(sA, LDSA_STRIDE, sB, LDSA_STRIDE, w2, b2, C1, C2, lane, wave);
  __syncthreads();
  // Max over K neighbors (values are post-ReLU, >= 0).
  for (int u = tid; u < G * C2; u += WGS) {
    int g = u / C2, c = u - g * C2;
    float m = 0.0f;
    for (int k = 0; k < K; ++k)
      m = fmaxf(m, (float)sB[(size_t)(g * K + k) * LDSA_STRIDE + c]);
    fout[((size_t)b * S + (s0 + g)) * C2 + c] = m;
  }
}

// ---------------------------------------------------------------------------
// Fused FP stage: 3-NN interpolate + concat -> MLP x2. 64 rows / workgroup.
// ---------------------------------------------------------------------------
__global__ __launch_bounds__(256) void k_fp_mlp(
    const float* __restrict__ unkf, int Nn,   // (B,Nn,256)
    const float* __restrict__ knf, int M,     // (B,M,256)
    const int* __restrict__ ki, const float* __restrict__ kw,  // (B,Nn,3)
    const _Float16* __restrict__ w0, const float* __restrict__ b0,
    const _Float16* __restrict__ w1, const float* __restrict__ b1,
    float* __restrict__ fout)                 // (B,Nn,256)
{
  __shared__ _Float16 sA[64 * FP_STR0];   // 512-wide input; reused (272-stride) for layer-1 out
  __shared__ _Float16 sB[64 * FP_STR1];   // 256-wide layer-0 out
  __shared__ int   s_ki[64][3];
  __shared__ float s_kw[64][3];
  const int b   = blockIdx.y;
  const int n0  = blockIdx.x * 64;
  const int tid = threadIdx.x, lane = tid & 31, wave = tid >> 5;
  if (tid < 64) {
    #pragma unroll
    for (int j = 0; j < 3; ++j) {
      s_ki[tid][j] = ki[((size_t)b * Nn + n0 + tid) * 3 + j];
      s_kw[tid][j] = kw[((size_t)b * Nn + n0 + tid) * 3 + j];
    }
  }
  __syncthreads();
  const float* Kb = knf + (size_t)b * M * 256;
  for (int u = tid; u < 64 * 512; u += WGS) {
    int r = u >> 9, c = u & 511;
    float v;
    if (c < 256) {
      v = Kb[(size_t)s_ki[r][0] * 256 + c] * s_kw[r][0]
        + Kb[(size_t)s_ki[r][1] * 256 + c] * s_kw[r][1]
        + Kb[(size_t)s_ki[r][2] * 256 + c] * s_kw[r][2];
    } else {
      v = unkf[((size_t)b * Nn + n0 + r) * 256 + (c - 256)];
    }
    sA[(size_t)r * FP_STR0 + c] = (_Float16)v;
  }
  __syncthreads();
  wmma_mlp_layer(sA, FP_STR0, sB, FP_STR1, w0, b0, 512, 256, lane, wave);
  __syncthreads();
  wmma_mlp_layer(sB, FP_STR1, sA, FP_STR1, w1, b1, 256, 256, lane, wave);
  __syncthreads();
  for (int u = tid; u < 64 * 256; u += WGS) {
    int r = u >> 8, c = u & 255;
    fout[((size_t)b * Nn + n0 + r) * 256 + c] = (float)sA[(size_t)r * FP_STR1 + c];
  }
}

// ---------------------------------------------------------------------------
// Final packing: features (B,256,1024) transposed ++ fp2_xyz ++ fp2_inds.
// ---------------------------------------------------------------------------
__global__ void k_finalize(const float* __restrict__ fp2o,   // (2,1024,256)
                           const float* __restrict__ xyz2,   // (2,1024,3)
                           const int* __restrict__ fpsi1,    // (2,2048)
                           float* __restrict__ out)
{
  int t = blockIdx.x * blockDim.x + threadIdx.x;
  const int NF = 2 * 256 * 1024;  // 524288
  const int NX = 2 * 1024 * 3;    // 6144
  const int NI = 2 * 1024;        // 2048
  if (t < NF) {
    int b = t >> 18, rem = t & 262143;
    int c = rem >> 10, n = rem & 1023;
    out[t] = fp2o[(((size_t)b << 10) + n) * 256 + c];
  } else if (t < NF + NX) {
    out[t] = xyz2[t - NF];
  } else if (t < NF + NX + NI) {
    int u = t - NF - NX;
    int b = u >> 10, n = u & 1023;
    out[t] = (float)fpsi1[b * 2048 + n];
  }
}

// ---------------------------------------------------------------------------
extern "C" void kernel_launch(void* const* d_in, const int* in_sizes, int n_in,
                              void* d_out, int out_size, void* d_ws, size_t ws_size,
                              hipStream_t stream)
{
  (void)n_in; (void)in_sizes; (void)out_size; (void)ws_size;
  const int B = 2, N0 = 20000;
  const float* pc = (const float*)d_in[0];

  // Weight tensors (d_in index, Cin, Cout) in MLP order.
  static const int widx[16]  = {1,3,5, 7,9,11, 13,15,17, 19,21,23, 25,27, 29,31};
  static const int bidx[16]  = {2,4,6, 8,10,12, 14,16,18, 20,22,24, 26,28, 30,32};
  static const int wCin[16]  = {3,64,64, 131,128,128, 259,128,128, 259,128,128, 512,256, 512,256};
  static const int wCout[16] = {64,64,128, 128,128,256, 128,128,256, 128,128,256, 256,256, 256,256};

  char* wsp = (char*)d_ws;
  size_t off = 0;
  auto alloc = [&](size_t bytes) -> char* {
    off = (off + 255) & ~(size_t)255;
    char* p = wsp + off;
    off += bytes;
    return p;
  };

  // Pack all MLP weights into f16 WMMA B-fragment order (re-done each call;
  // deterministic).
  _Float16* W16[16];
  const float* bias[16];
  for (int i = 0; i < 16; ++i) {
    const int nKt = (wCin[i] + 31) / 32;
    const int nCt = wCout[i] / 16;
    const int total = nKt * nCt * 512;   // 32 lanes * 16 halves per fragment
    W16[i] = (_Float16*)alloc((size_t)total * sizeof(_Float16));
    k_pack_w<<<dim3((total + 255) / 256), dim3(256), 0, stream>>>(
        (const float*)d_in[widx[i]], W16[i], wCin[i], wCout[i], total);
    bias[i] = (const float*)d_in[bidx[i]];
  }

  float* dist = (float*)alloc((size_t)B * N0 * 4);
  int*   fi1  = (int*)alloc((size_t)B * 2048 * 4);
  int*   fi2  = (int*)alloc((size_t)B * 1024 * 4);
  int*   fi3  = (int*)alloc((size_t)B * 512 * 4);
  int*   fi4  = (int*)alloc((size_t)B * 256 * 4);
  float* x1   = (float*)alloc((size_t)B * 2048 * 3 * 4);
  float* x2   = (float*)alloc((size_t)B * 1024 * 3 * 4);
  float* x3   = (float*)alloc((size_t)B * 512 * 3 * 4);
  float* x4   = (float*)alloc((size_t)B * 256 * 3 * 4);
  int*   nb1  = (int*)alloc((size_t)B * 2048 * 64 * 4);
  int*   nb2  = (int*)alloc((size_t)B * 1024 * 32 * 4);
  int*   nb3  = (int*)alloc((size_t)B * 512 * 16 * 4);
  int*   nb4  = (int*)alloc((size_t)B * 256 * 16 * 4);
  float* f1   = (float*)alloc((size_t)B * 2048 * 128 * 4);
  float* f2   = (float*)alloc((size_t)B * 1024 * 256 * 4);
  float* f3   = (float*)alloc((size_t)B * 512 * 256 * 4);
  float* f4   = (float*)alloc((size_t)B * 256 * 256 * 4);
  int*   ki1  = (int*)alloc((size_t)B * 512 * 3 * 4);
  float* kw1  = (float*)alloc((size_t)B * 512 * 3 * 4);
  int*   ki2  = (int*)alloc((size_t)B * 1024 * 3 * 4);
  float* kw2  = (float*)alloc((size_t)B * 1024 * 3 * 4);
  float* fp1o = (float*)alloc((size_t)B * 512 * 256 * 4);
  float* fp2o = (float*)alloc((size_t)B * 1024 * 256 * 4);

  // ---- SA1: N=20000 -> S=2048, K=64, r=0.04, MLP 3->64->64->128
  k_fps<<<dim3(B), dim3(1024), 0, stream>>>(pc, N0, 2048, fi1, dist);
  k_gather_xyz<<<dim3((B*2048+255)/256), dim3(256), 0, stream>>>(pc, N0, fi1, 2048, x1, B*2048);
  k_ball_query<<<dim3((B*2048+255)/256), dim3(256), 0, stream>>>(pc, N0, x1, 2048, 64, 0.04f*0.04f, nb1, B*2048);
  k_sa_mlp<<<dim3(2048, B), dim3(256), 0, stream>>>(pc, N0, x1, 2048, nb1, 64, nullptr, 0, 25.0f,
      W16[0], bias[0], 64, W16[1], bias[1], 64, W16[2], bias[2], 128, f1);

  // ---- SA2: 2048 -> 1024, K=32, r=0.1, MLP 131->128->128->256
  k_fps<<<dim3(B), dim3(1024), 0, stream>>>(x1, 2048, 1024, fi2, dist);
  k_gather_xyz<<<dim3((B*1024+255)/256), dim3(256), 0, stream>>>(x1, 2048, fi2, 1024, x2, B*1024);
  k_ball_query<<<dim3((B*1024+255)/256), dim3(256), 0, stream>>>(x1, 2048, x2, 1024, 32, 0.1f*0.1f, nb2, B*1024);
  k_sa_mlp<<<dim3(512, B), dim3(256), 0, stream>>>(x1, 2048, x2, 1024, nb2, 32, f1, 128, 10.0f,
      W16[3], bias[3], 128, W16[4], bias[4], 128, W16[5], bias[5], 256, f2);

  // ---- SA3: 1024 -> 512, K=16, r=0.2, MLP 259->128->128->256
  k_fps<<<dim3(B), dim3(1024), 0, stream>>>(x2, 1024, 512, fi3, dist);
  k_gather_xyz<<<dim3((B*512+255)/256), dim3(256), 0, stream>>>(x2, 1024, fi3, 512, x3, B*512);
  k_ball_query<<<dim3((B*512+255)/256), dim3(256), 0, stream>>>(x2, 1024, x3, 512, 16, 0.2f*0.2f, nb3, B*512);
  k_sa_mlp<<<dim3(128, B), dim3(256), 0, stream>>>(x2, 1024, x3, 512, nb3, 16, f2, 256, 5.0f,
      W16[6], bias[6], 128, W16[7], bias[7], 128, W16[8], bias[8], 256, f3);

  // ---- SA4: 512 -> 256, K=16, r=0.3, MLP 259->128->128->256
  k_fps<<<dim3(B), dim3(1024), 0, stream>>>(x3, 512, 256, fi4, dist);
  k_gather_xyz<<<dim3((B*256+255)/256), dim3(256), 0, stream>>>(x3, 512, fi4, 256, x4, B*256);
  k_ball_query<<<dim3((B*256+255)/256), dim3(256), 0, stream>>>(x3, 512, x4, 256, 16, 0.3f*0.3f, nb4, B*256);
  k_sa_mlp<<<dim3(64, B), dim3(256), 0, stream>>>(x3, 512, x4, 256, nb4, 16, f3, 256, 1.0f/0.3f,
      W16[9], bias[9], 128, W16[10], bias[10], 128, W16[11], bias[11], 256, f4);

  // ---- FP1: unknown=sa3 (512), known=sa4 (256); MLP 512->256->256
  k_knn3<<<dim3((B*512+255)/256), dim3(256), 0, stream>>>(x3, 512, x4, 256, ki1, kw1, B*512);
  k_fp_mlp<<<dim3(8, B), dim3(256), 0, stream>>>(f3, 512, f4, 256, ki1, kw1,
      W16[12], bias[12], W16[13], bias[13], fp1o);

  // ---- FP2: unknown=sa2 (1024), known=sa3 (512); MLP 512->256->256
  k_knn3<<<dim3((B*1024+255)/256), dim3(256), 0, stream>>>(x2, 1024, x3, 512, ki2, kw2, B*1024);
  k_fp_mlp<<<dim3(16, B), dim3(256), 0, stream>>>(f2, 1024, fp1o, 512, ki2, kw2,
      W16[14], bias[14], W16[15], bias[15], fp2o);

  // ---- outputs
  k_finalize<<<dim3((532480 + 255) / 256), dim3(256), 0, stream>>>(fp2o, x2, fi1, (float*)d_out);
}
